// AttentionBlock_17927193494079
// MI455X (gfx1250) — compile-verified, matrix operand order checked by reading, not run on hardware
//
#include <hip/hip_runtime.h>
#include <hip/hip_bf16.h>

// ---------------- problem constants ----------------
constexpr int BATCH = 4;
constexpr int HID   = 1024;
constexpr int T     = 2048;
constexpr int D     = 1024;
#define NEG_INF_VAL (-32767.0f)

typedef __attribute__((ext_vector_type(16))) _Float16 v16h;
typedef __attribute__((ext_vector_type(8)))  _Float16 v8h;
typedef __attribute__((ext_vector_type(8)))  float    v8f;

// CDNA5 async global->LDS copy path (ASYNCcnt-tracked DMA, no VGPR staging).
// Guarded: device pass only, and only if the toolchain declares the builtins.
// Probe (round 2 stderr) showed param types: (int __vector_size(16) AS(1)*,
//  int __vector_size(16) AS(3)*, imm int offset, imm int cpol).
#if defined(__AMDGCN__) &&                                            \
    __has_builtin(__builtin_amdgcn_global_load_async_to_lds_b128) &&  \
    __has_builtin(__builtin_amdgcn_s_wait_asynccnt)
#define USE_ASYNC_LDS 1
typedef int async_b128_t __attribute__((vector_size(4 * sizeof(int))));
typedef __attribute__((address_space(1))) async_b128_t* gas_b128_ptr;
typedef __attribute__((address_space(3))) async_b128_t* lds_b128_ptr;
#else
#define USE_ASYNC_LDS 0
#endif

// ---------------- WMMA GEMM: C[m,n] = sum_k A[m,k] * W[n,k] (+epilogue) ----------------
// A: [M,K] f16 row-major, W: [N,K] f16 row-major (both K-contiguous -> matches
// CDNA5 WMMA A/B fragment layouts: each lane holds one M-row / N-col stripe).
constexpr int BM  = 128;
constexpr int BN  = 128;
constexpr int BK  = 32;
constexpr int LDT = 40;   // LDS row stride in halves (32 + 8 pad)

// A-fragment (16x32 f16): lane l -> row (l&15); K chunks [kb,kb+8) and [kb+16,kb+24),
// kb = (l>=16 ? 8 : 0).  B-fragment mirrors with N instead of M.
__device__ __forceinline__ v16h make_frag(const _Float16* row, int kb) {
  v8h lo = *(const v8h*)(row + kb);
  v8h hi = *(const v8h*)(row + kb + 16);
  v16h r;
#pragma unroll
  for (int i = 0; i < 8; ++i) { r[i] = lo[i]; r[i + 8] = hi[i]; }
  return r;
}

// MODE 0: out16 = out32 = C + bias          (embedding projection: need f32 residual copy)
// MODE 1: out16 = C + bias                  (K / Q / V projections)
// MODE 2: out32 = causal-masked C (s<=t)    (logits, M=N=T)
// MODE 3: outT[n*M+m] = C + resid[m*N+n]    (probs@V, residual add, transposed store)
template <int MODE>
__global__ __launch_bounds__(256) void gemm16_kernel(
    const _Float16* __restrict__ A, const _Float16* __restrict__ W,
    const float* __restrict__ bias,
    _Float16* __restrict__ out16, float* __restrict__ out32,
    const float* __restrict__ resid, float* __restrict__ outT,
    int M, int N, int K, long aStrideB, long wStrideB) {
  __shared__ __align__(16) _Float16 sA[2][BM * LDT];
  __shared__ __align__(16) _Float16 sB[2][BN * LDT];

  const int  b = blockIdx.z;
  A += (long)b * aStrideB;
  W += (long)b * wStrideB;
  const long mnStride = (long)M * N;

  const int mBase = blockIdx.y * BM;
  const int nBase = blockIdx.x * BN;
  const int tid   = threadIdx.x;

  if (MODE == 2) {
    // tile entirely above the causal diagonal -> pure NEG_INF fill, skip GEMM
    if (nBase > mBase + (BM - 1)) {
      float* o = out32 + b * mnStride;
      for (int idx = tid; idx < BM * BN; idx += 256) {
        int r = idx >> 7, c = idx & (BN - 1);
        o[(long)(mBase + r) * N + (nBase + c)] = NEG_INF_VAL;
      }
      return;
    }
  }

  const int lane  = tid & 31;
  const int wave  = tid >> 5;             // 8 waves
  const int waveM = (wave & 1) * 64;      // 2 waves along M, 64 rows each
  const int waveN = (wave >> 1) * 32;     // 4 waves along N, 32 cols each
  const int lr    = lane & 15;
  const int kb    = (lane >> 4) * 8;      // fragment K base per lane group
  const int hi8   = (lane >> 4) * 8;      // C/D row offset for upper lane half

  // cooperative tile staging: thread -> 8 halves (16B), rows r0 and r0+64
  const int r0 = tid >> 2;                // 0..63
  const int c0 = (tid & 3) * 8;           // 0,8,16,24

  v8f acc[4][2] = {};

  auto compute_step = [&](const _Float16* tA, const _Float16* tB) {
    v16h af[4], bf[2];
#pragma unroll
    for (int mt = 0; mt < 4; ++mt)
      af[mt] = make_frag(&tA[(waveM + mt * 16 + lr) * LDT], kb);
#pragma unroll
    for (int nt = 0; nt < 2; ++nt)
      bf[nt] = make_frag(&tB[(waveN + nt * 16 + lr) * LDT], kb);
#pragma unroll
    for (int mt = 0; mt < 4; ++mt)
#pragma unroll
      for (int nt = 0; nt < 2; ++nt)
        acc[mt][nt] = __builtin_amdgcn_wmma_f32_16x16x32_f16(
            false, af[mt], false, bf[nt], (short)0, acc[mt][nt], false, false);
  };

#if USE_ASYNC_LDS
  // ---- double-buffered, ASYNCcnt-tracked DMA staging (4 b128 ops/thread/slice) ----
  auto stage = [&](int k0, int bufi) {
    __builtin_amdgcn_global_load_async_to_lds_b128(
        (gas_b128_ptr)(A + (long)(mBase + r0) * K + k0 + c0),
        (lds_b128_ptr)(&sA[bufi][r0 * LDT + c0]), 0, 0);
    __builtin_amdgcn_global_load_async_to_lds_b128(
        (gas_b128_ptr)(A + (long)(mBase + r0 + 64) * K + k0 + c0),
        (lds_b128_ptr)(&sA[bufi][(r0 + 64) * LDT + c0]), 0, 0);
    __builtin_amdgcn_global_load_async_to_lds_b128(
        (gas_b128_ptr)(W + (long)(nBase + r0) * K + k0 + c0),
        (lds_b128_ptr)(&sB[bufi][r0 * LDT + c0]), 0, 0);
    __builtin_amdgcn_global_load_async_to_lds_b128(
        (gas_b128_ptr)(W + (long)(nBase + r0 + 64) * K + k0 + c0),
        (lds_b128_ptr)(&sB[bufi][(r0 + 64) * LDT + c0]), 0, 0);
  };
  stage(0, 0);
  int bufi = 0;
  for (int k0 = 0; k0 < K; k0 += BK, bufi ^= 1) {
    if (k0 + BK < K) {
      // overlap next-slice DMA with this slice's WMMAs; buffer bufi^1 was last
      // read an iteration ago and a barrier has passed since.
      stage(k0 + BK, bufi ^ 1);
      __builtin_amdgcn_s_wait_asynccnt(4);  // our 4 current-slice ops complete (in-order)
    } else {
      __builtin_amdgcn_s_wait_asynccnt(0);
    }
    __syncthreads();                        // all waves' DMA for bufi visible
    compute_step(sA[bufi], sB[bufi]);
    __syncthreads();                        // all waves done reading bufi
  }
#else
  // ---- fallback: register-staged LDS tiles ----
  for (int k0 = 0; k0 < K; k0 += BK) {
    const v8h ga0 = *(const v8h*)(A + (long)(mBase + r0)      * K + k0 + c0);
    const v8h ga1 = *(const v8h*)(A + (long)(mBase + r0 + 64) * K + k0 + c0);
    const v8h gb0 = *(const v8h*)(W + (long)(nBase + r0)      * K + k0 + c0);
    const v8h gb1 = *(const v8h*)(W + (long)(nBase + r0 + 64) * K + k0 + c0);
    if (k0 + BK < K) {  // global_prefetch_b8 of next K-slice
      __builtin_prefetch(A + (long)(mBase + r0) * K + k0 + BK + c0, 0, 1);
      __builtin_prefetch(W + (long)(nBase + r0) * K + k0 + BK + c0, 0, 1);
    }
    __syncthreads();
    *(v8h*)(&sA[0][(r0)      * LDT + c0]) = ga0;
    *(v8h*)(&sA[0][(r0 + 64) * LDT + c0]) = ga1;
    *(v8h*)(&sB[0][(r0)      * LDT + c0]) = gb0;
    *(v8h*)(&sB[0][(r0 + 64) * LDT + c0]) = gb1;
    __syncthreads();
    compute_step(sA[0], sB[0]);
  }
#endif

  // epilogue; C/D layout: vgpr v, lane -> row = v + 8*(lane>=16), col = lane&15
#pragma unroll
  for (int mt = 0; mt < 4; ++mt) {
#pragma unroll
    for (int nt = 0; nt < 2; ++nt) {
      const int n = nBase + waveN + nt * 16 + lr;
#pragma unroll
      for (int v = 0; v < 8; ++v) {
        const int m = mBase + waveM + mt * 16 + hi8 + v;
        float val = acc[mt][nt][v];
        if (MODE == 0) {
          val += bias[n];
          out16[b * mnStride + (long)m * N + n] = (_Float16)val;
          out32[b * mnStride + (long)m * N + n] = val;
        } else if (MODE == 1) {
          val += bias[n];
          out16[b * mnStride + (long)m * N + n] = (_Float16)val;
        } else if (MODE == 2) {
          out32[b * mnStride + (long)m * N + n] = (n <= m) ? val : NEG_INF_VAL;
        } else {
          val += resid[b * mnStride + (long)m * N + n];
          outT[b * mnStride + (long)n * M + m] = val;
        }
      }
    }
  }
}

// ---------------- tiled transpose (+convert to f16): out[c][r] = in[r][c] ----------------
template <typename Tin>
__global__ __launch_bounds__(256) void transpose_to_f16_kernel(
    const Tin* __restrict__ in, _Float16* __restrict__ out, int R, int C) {
  __shared__ float tile[32][33];
  const int b = blockIdx.z;
  in  += (long)b * R * C;
  out += (long)b * R * C;
  const int tx = threadIdx.x, ty = threadIdx.y;  // 32 x 8
  const int c = blockIdx.x * 32 + tx;
#pragma unroll
  for (int j = 0; j < 32; j += 8) {
    const int r = blockIdx.y * 32 + ty + j;
    tile[ty + j][tx] = (float)in[(long)r * C + c];
  }
  __syncthreads();
#pragma unroll
  for (int j = 0; j < 32; j += 8) {
    const int orow = blockIdx.x * 32 + ty + j;   // out row = original column
    const int ocol = blockIdx.y * 32 + tx;       // out col = original row
    out[(long)orow * R + ocol] = (_Float16)tile[tx][ty + j];
  }
}

// ---------------- f32 -> f16 elementwise convert (weights) ----------------
__global__ __launch_bounds__(256) void cvt_f16_kernel(
    const float* __restrict__ in, _Float16* __restrict__ out, int n) {
  int i = blockIdx.x * 256 + threadIdx.x;
  if (i < n) out[i] = (_Float16)in[i];
}

// ---------------- column softmax over axis t (query axis), * 1/sqrt(D) ----------------
__global__ __launch_bounds__(256) void softmax_col_kernel(
    const float* __restrict__ logits, _Float16* __restrict__ probs, int n) {
  const int b = blockIdx.z;
  logits += (long)b * n * n;
  probs  += (long)b * n * n;
  const int s = blockIdx.x * 256 + threadIdx.x;  // column index (coalesced)
  float mx = -3.0e38f, l = 0.0f;
  for (int t = 0; t < n; ++t) {                  // online max/sum down the column
    float v  = logits[(long)t * n + s];
    float nm = fmaxf(mx, v);
    l  = l * __expf(mx - nm) + __expf(v - nm);
    mx = nm;
  }
  const float inv = 0.03125f / l;                // softmax / sqrt(1024)
  for (int t = 0; t < n; ++t) {
    float v = logits[(long)t * n + s];
    probs[(long)t * n + s] = (_Float16)(__expf(v - mx) * inv);
  }
}

// ---------------- driver ----------------
extern "C" void kernel_launch(void* const* d_in, const int* in_sizes, int n_in,
                              void* d_out, int out_size, void* d_ws, size_t ws_size,
                              hipStream_t stream) {
  (void)in_sizes; (void)n_in; (void)out_size; (void)ws_size;
  const float* minibatch = (const float*)d_in[0];
  const float* emb_w   = (const float*)d_in[1];
  const float* emb_b   = (const float*)d_in[2];
  const float* key_w   = (const float*)d_in[3];
  const float* key_b   = (const float*)d_in[4];
  const float* query_w = (const float*)d_in[5];
  const float* query_b = (const float*)d_in[6];
  const float* value_w = (const float*)d_in[7];
  const float* value_b = (const float*)d_in[8];
  float* out = (float*)d_out;

  char* p = (char*)d_ws;
  auto alloc = [&](size_t bytes) -> char* {
    char* r = p;
    p += (bytes + 255) & ~(size_t)255;
    return r;
  };
  _Float16* xT    = (_Float16*)alloc((size_t)BATCH * T * HID * 2); // [B,T,HID] f16
  _Float16* embw16= (_Float16*)alloc((size_t)D * HID * 2);
  _Float16* keyw16= (_Float16*)alloc((size_t)D * D * 2);
  _Float16* qryw16= (_Float16*)alloc((size_t)D * D * 2);
  _Float16* valw16= (_Float16*)alloc((size_t)D * D * 2);
  float*    m32   = (float*)   alloc((size_t)BATCH * T * D * 4);   // residual copy
  _Float16* m16   = (_Float16*)alloc((size_t)BATCH * T * D * 2);
  _Float16* k16   = (_Float16*)alloc((size_t)BATCH * T * D * 2);
  _Float16* q16   = (_Float16*)alloc((size_t)BATCH * T * D * 2);
  _Float16* v16b  = (_Float16*)alloc((size_t)BATCH * T * D * 2);
  _Float16* vT16  = (_Float16*)alloc((size_t)BATCH * D * T * 2);   // [B,D,T]
  float*    lgt32 = (float*)   alloc((size_t)BATCH * T * T * 4);
  _Float16* pr16  = (_Float16*)alloc((size_t)BATCH * T * T * 2);

  const dim3 tb(32, 8);

  // 1) x = transpose(minibatch): [HID,T] -> [T,HID] f16
  transpose_to_f16_kernel<float><<<dim3(T / 32, HID / 32, BATCH), tb, 0, stream>>>(
      minibatch, xT, HID, T);

  // 2) weights f32 -> f16 (K-contiguous already)
  const int wn = D * HID;
  cvt_f16_kernel<<<(wn + 255) / 256, 256, 0, stream>>>(emb_w,   embw16, wn);
  cvt_f16_kernel<<<(wn + 255) / 256, 256, 0, stream>>>(key_w,   keyw16, wn);
  cvt_f16_kernel<<<(wn + 255) / 256, 256, 0, stream>>>(query_w, qryw16, wn);
  cvt_f16_kernel<<<(wn + 255) / 256, 256, 0, stream>>>(value_w, valw16, wn);

  // 3) m = x @ emb_w^T + emb_b  (f16 for reuse + f32 residual)
  const dim3 gProj(D / BN, T / BM, BATCH);
  gemm16_kernel<0><<<gProj, 256, 0, stream>>>(
      xT, embw16, emb_b, m16, m32, nullptr, nullptr, T, D, HID, (long)T * HID, 0);

  // 4) K / Q / V projections
  gemm16_kernel<1><<<gProj, 256, 0, stream>>>(
      m16, keyw16, key_b, k16, nullptr, nullptr, nullptr, T, D, D, (long)T * D, 0);
  gemm16_kernel<1><<<gProj, 256, 0, stream>>>(
      m16, qryw16, query_b, q16, nullptr, nullptr, nullptr, T, D, D, (long)T * D, 0);
  gemm16_kernel<1><<<gProj, 256, 0, stream>>>(
      m16, valw16, value_b, v16b, nullptr, nullptr, nullptr, T, D, D, (long)T * D, 0);

  // 5) V^T so probs@V has a K-contiguous B operand
  transpose_to_f16_kernel<_Float16><<<dim3(D / 32, T / 32, BATCH), tb, 0, stream>>>(
      v16b, vT16, T, D);

  // 6) logits = Q K^T with causal mask (tile-skip above diagonal)
  gemm16_kernel<2><<<dim3(T / BN, T / BM, BATCH), 256, 0, stream>>>(
      q16, k16, nullptr, nullptr, lgt32, nullptr, nullptr,
      T, T, D, (long)T * D, (long)T * D);

  // 7) softmax over query axis (axis=1), scaled by 1/sqrt(D), to f16
  softmax_col_kernel<<<dim3(T / 256, 1, BATCH), 256, 0, stream>>>(lgt32, pr16, T);

  // 8) read = probs @ V; out[b,d,t] = read[b,t,d] + m[b,t,d]
  gemm16_kernel<3><<<dim3(D / BN, T / BM, BATCH), 256, 0, stream>>>(
      pr16, vT16, nullptr, nullptr, nullptr, m32, out,
      T, D, T, (long)T * T, (long)D * T);
}
